// DynamicVFE_55327768707746
// MI455X (gfx1250) — matrix operand order repaired and checked by reading, compiled, not run on hardware
//
#include <hip/hip_runtime.h>

typedef __attribute__((ext_vector_type(16))) _Float16 v16h;
typedef __attribute__((ext_vector_type(8)))  float    v8f;
typedef __attribute__((ext_vector_type(4)))  int      v4i_;

#define NPTS   500000
#define CZg    1
#define CYg    400
#define CXg    352
#define CLg    281600            // 2*1*400*352
#define CH     64
#define CIN    11
#define VXc    0.2f
#define VYc    0.2f
#define VZc    4.0f
#define XOFFc  0.1f              // VX/2 + X0
#define YOFFc  (-39.9f)          // VY/2 + Y0
#define ZOFFc  (-1.0f)           // VZ/2 + Z0

// workspace layout (bytes, 256-aligned)
#define OFF_IDX   0UL                         // N * 4        = 2,000,000
#define OFF_CNT   2000128UL                   // CL * 4       = 1,126,400
#define OFF_VSUM  3126528UL                   // CL * 3 * 4   = 3,379,200
#define OFF_VMAX  6505728UL                   // CL * 64 * 4  = 72,089,600
#define OFF_PF1   78595328UL                  // N * 64 * 2   = 64,000,000

// ---- gfx1250 async global->LDS (ASYNCcnt-tracked), guarded so compile never breaks ----
#if defined(__has_builtin)
#  if __has_builtin(__builtin_amdgcn_global_load_async_to_lds_b128)
#    define HAVE_ASYNC_LDS 1
#  endif
#endif
#ifndef HAVE_ASYNC_LDS
#  define HAVE_ASYNC_LDS 0
#endif

// builtin proto (from hipcc diagnostic): (AS1 v4i*, AS3 v4i*, imm int, imm int)
#define AS1_V4I(p) ((__attribute__((address_space(1))) v4i_*)(unsigned long long)(const void*)(p))
#define AS3_V4I(p) ((__attribute__((address_space(3))) v4i_*)(unsigned int)(unsigned long long)(const void*)(p))

// ---------------- K0: zero-init accumulators + output ----------------
__global__ void k0_init(float* cnt, float* vsum, int* vmax1, int* out) {
    long i = (long)blockIdx.x * blockDim.x + threadIdx.x;
    long stride = (long)gridDim.x * blockDim.x;
    for (long j = i; j < CLg; j += stride) cnt[j] = 0.0f;
    for (long j = i; j < 3L * CLg; j += stride) vsum[j] = 0.0f;
    for (long j = i; j < 64L * CLg; j += stride) { vmax1[j] = 0; out[j] = 0; }
}

// ---------------- K1: voxel index + count/sum scatter ----------------
__global__ void k1_scatter(const float* __restrict__ feat, const int* __restrict__ coors,
                           int* __restrict__ idxArr, float* cnt, float* vsum) {
    int i = blockIdx.x * blockDim.x + threadIdx.x;
    if (i >= NPTS) return;
    int b = coors[i * 4 + 0], z = coors[i * 4 + 1];
    int y = coors[i * 4 + 2], x = coors[i * 4 + 3];
    unsigned idx = (unsigned)(((b * CZg + z) * CYg + y) * CXg + x);
    idxArr[i] = (int)idx;
    atomicAdd(&cnt[idx], 1.0f);
    atomicAdd(&vsum[idx * 3u + 0], feat[i * 4 + 0]);
    atomicAdd(&vsum[idx * 3u + 1], feat[i * 4 + 1]);
    atomicAdd(&vsum[idx * 3u + 2], feat[i * 4 + 2]);
}

// ---------------- K2: 11-dim features -> MLP1 -> pf(fp16) + segment-max ----------------
__global__ void k2_mlp1(const float* __restrict__ feat, const int* __restrict__ coors,
                        const int* __restrict__ idxArr, const float* __restrict__ cnt,
                        const float* __restrict__ vsum, const float* __restrict__ W1,
                        const float* __restrict__ scale1, const float* __restrict__ shift1,
                        _Float16* __restrict__ pf1, int* __restrict__ vmax1) {
    __shared__ float sW1[CIN * CH];
    __shared__ float sSc[CH], sSh[CH];
    for (int t = threadIdx.x; t < CIN * CH; t += blockDim.x) sW1[t] = W1[t];
    for (int t = threadIdx.x; t < CH; t += blockDim.x) { sSc[t] = scale1[t]; sSh[t] = shift1[t]; }
    __syncthreads();

    int i = blockIdx.x * blockDim.x + threadIdx.x;
    if (i >= NPTS) return;
    float fx = feat[i * 4 + 0], fy = feat[i * 4 + 1];
    float fz = feat[i * 4 + 2], fr = feat[i * 4 + 3];
    int z = coors[i * 4 + 1], y = coors[i * 4 + 2], x = coors[i * 4 + 3];
    unsigned vi = (unsigned)idxArr[i];
    float c = fmaxf(cnt[vi], 1.0f);
    float f[CIN];
    f[0] = fx; f[1] = fy; f[2] = fz; f[3] = fr;
    f[4] = fx - vsum[vi * 3u + 0] / c;
    f[5] = fy - vsum[vi * 3u + 1] / c;
    f[6] = fz - vsum[vi * 3u + 2] / c;
    f[7] = fx - ((float)x * VXc + XOFFc);
    f[8] = fy - ((float)y * VYc + YOFFc);
    f[9] = fz - ((float)z * VZc + ZOFFc);
    f[10] = sqrtf(fx * fx + fy * fy + fz * fz);

    unsigned pfBase = (unsigned)i * 64u;
    unsigned vmBase = vi * 64u;
    for (int h = 0; h < CH; ++h) {
        float acc = 0.0f;
#pragma unroll
        for (int k = 0; k < CIN; ++k) acc = fmaf(f[k], sW1[k * CH + h], acc);
        float v = fmaxf(acc * sSc[h] + sSh[h], 0.0f);
        pf1[pfBase + h] = (_Float16)v;
        atomicMax(&vmax1[vmBase + h], __float_as_int(v));   // v >= 0: int order == float order
    }
}

// ---------------- K3: MLP2 via v_wmma_f32_16x16x32_f16 + segment-max ----------------
#define K3_WAVES   4
#define TILES      (NPTS / 16)     // 31250, exact
#define W2T_STRIDE 130             // 128 halves + 2 pad (bank-conflict mitigation)

// stage the pf half (K=0..63) of a 16-point A tile: 16 rows x 128B, global -> LDS
__device__ __forceinline__ void stage_pf(const _Float16* __restrict__ pf1,
                                         _Float16* A, int p0, int lane) {
#if HAVE_ASYNC_LDS
    // 4 async-DMA instructions (16B per lane each); tracked by ASYNCcnt
    for (int t = lane; t < 128; t += 32) {
        int m = t >> 3, q = t & 7;
        __builtin_amdgcn_global_load_async_to_lds_b128(
            AS1_V4I(pf1 + (unsigned)(p0 + m) * 64u + q * 8),
            AS3_V4I(A + m * 128 + q * 8),
            0, 0);
    }
#else
    for (int t = lane; t < 512; t += 32) {
        int m = t >> 5, kd = t & 31;
        ((unsigned int*)A)[m * 64 + kd] =
            ((const unsigned int*)pf1)[(unsigned)(p0 + m) * 32u + kd];
    }
#endif
}

__global__ void k3_mlp2(const _Float16* __restrict__ pf1, const float* __restrict__ vmax1f,
                        const int* __restrict__ idxArr, const float* __restrict__ W2,
                        const float* __restrict__ scale2, const float* __restrict__ shift2,
                        int* __restrict__ outI) {
    __shared__ __align__(16) _Float16 sW2T[64 * W2T_STRIDE];   // W2 transposed [n][k], fp16
    __shared__ float sSc[64], sSh[64];
    __shared__ __align__(16) _Float16 sA[K3_WAVES][16 * 128];  // per-wave A tile (16 pts x 128 feat)

    for (int t = threadIdx.x; t < 128 * 64; t += blockDim.x) {
        int k = t >> 6, n = t & 63;
        sW2T[n * W2T_STRIDE + k] = (_Float16)W2[t];
    }
    for (int t = threadIdx.x; t < 64; t += blockDim.x) { sSc[t] = scale2[t]; sSh[t] = shift2[t]; }
    __syncthreads();

    const int wave = threadIdx.x >> 5;
    const int lane = threadIdx.x & 31;
    const int hi   = lane >> 4;       // 0: lanes 0-15, 1: lanes 16-31
    const int lrow = lane & 15;       // M (for A/C) or N (for B/C) index
    _Float16* A = sA[wave];

    int gwave   = blockIdx.x * K3_WAVES + wave;
    int gstride = gridDim.x * K3_WAVES;

    // prologue: kick off the DMA for this wave's first tile
    if (gwave < TILES) stage_pf(pf1, A, gwave * 16, lane);

    for (int tile = gwave; tile < TILES; tile += gstride) {
        int p0 = tile * 16;

        // vmax part (K = 64..127): gather by voxel, f32 -> f16 (disjoint from async pf region)
        for (int t = lane; t < 1024; t += 32) {
            int m = t >> 6, k = t & 63;
            unsigned vox = (unsigned)idxArr[p0 + m];
            A[m * 128 + 64 + k] = (_Float16)vmax1f[vox * 64u + k];
        }
#if HAVE_ASYNC_LDS
#  if __has_builtin(__builtin_amdgcn_s_wait_asynccnt)
        __builtin_amdgcn_s_wait_asynccnt(0);
#  else
        asm volatile("s_wait_asynccnt 0x0" ::: "memory");
#  endif
#endif
        __builtin_amdgcn_wave_barrier();
        asm volatile("s_wait_dscnt 0x0" ::: "memory");   // LDS in-order per wave; wait before reads

        // A fragments: ISA 16-bit A 16x32 layout. lane(M=lrow), VGPR v:
        //   v<4 : K = 8*hi + 2*v + {0,1};  v>=4 : K = 16 + 8*hi + 2*(v-4) + {0,1}
        v16h Af[4];
#pragma unroll
        for (int ks = 0; ks < 4; ++ks) {
            v16h a;
#pragma unroll
            for (int j = 0; j < 8; ++j) {
                int kb = ks * 32 + ((j & 4) << 2) + 8 * hi + 2 * (j & 3);
                a[2 * j]     = A[lrow * 128 + kb];
                a[2 * j + 1] = A[lrow * 128 + kb + 1];
            }
            Af[ks] = a;
        }
        // A data now lives in registers; drain ds reads, then overlap next tile's DMA
        // with this tile's WMMA + atomic epilogue (async ops are unordered vs DS ops,
        // so the explicit dscnt drain prevents the DMA racing the fragment loads).
        asm volatile("s_wait_dscnt 0x0" ::: "memory");
        __builtin_amdgcn_wave_barrier();
        {
            int ntile = tile + gstride;
            if (ntile < TILES) stage_pf(pf1, A, ntile * 16, lane);
        }

#pragma unroll
        for (int nt = 0; nt < 4; ++nt) {
            v8f c = {};
#pragma unroll
            for (int ks = 0; ks < 4; ++ks) {
                // B 32x16 layout: lane(N=lrow), VGPR v: v<4: K = 8*hi + 2*v + {0,1}; v>=4: K = 16 + 8*hi + 2*(v-4) + {0,1}
                v16h b;
#pragma unroll
                for (int j = 0; j < 8; ++j) {
                    int kb = ks * 32 + ((j & 4) << 2) + 8 * hi + 2 * (j & 3);
                    const _Float16* bp = &sW2T[(nt * 16 + lrow) * W2T_STRIDE + kb];
                    b[2 * j]     = bp[0];
                    b[2 * j + 1] = bp[1];
                }
                c = __builtin_amdgcn_wmma_f32_16x16x32_f16(
                        false, Af[ks], false, b, (short)0, c, false, false);
            }
            // C layout: VGPR r -> M = r + 8*hi, N = lrow (+ nt*16)
            int n = nt * 16 + lrow;
            float sc = sSc[n], sh = sSh[n];
#pragma unroll
            for (int r = 0; r < 8; ++r) {
                int m = r + 8 * hi;
                float val = fmaxf(c[r] * sc + sh, 0.0f);
                unsigned off = (unsigned)idxArr[p0 + m] * 64u + (unsigned)n;  // 32-bit voffset
                atomicMax(&outI[off], __float_as_int(val));
            }
        }
    }
}

extern "C" void kernel_launch(void* const* d_in, const int* in_sizes, int n_in,
                              void* d_out, int out_size, void* d_ws, size_t ws_size,
                              hipStream_t stream) {
    const float* feat   = (const float*)d_in[0];
    const int*   coors  = (const int*)  d_in[1];
    const float* W1     = (const float*)d_in[2];
    const float* scale1 = (const float*)d_in[3];
    const float* shift1 = (const float*)d_in[4];
    const float* W2     = (const float*)d_in[5];
    const float* scale2 = (const float*)d_in[6];
    const float* shift2 = (const float*)d_in[7];

    char* ws = (char*)d_ws;
    int*      idxArr = (int*)     (ws + OFF_IDX);
    float*    cnt    = (float*)   (ws + OFF_CNT);
    float*    vsum   = (float*)   (ws + OFF_VSUM);
    int*      vmax1  = (int*)     (ws + OFF_VMAX);
    _Float16* pf1    = (_Float16*)(ws + OFF_PF1);

    k0_init<<<2048, 256, 0, stream>>>(cnt, vsum, vmax1, (int*)d_out);
    k1_scatter<<<(NPTS + 255) / 256, 256, 0, stream>>>(feat, coors, idxArr, cnt, vsum);
    k2_mlp1<<<(NPTS + 255) / 256, 256, 0, stream>>>(feat, coors, idxArr, cnt, vsum,
                                                    W1, scale1, shift1, pf1, vmax1);
    k3_mlp2<<<1024, 128, 0, stream>>>(pf1, (const float*)vmax1, idxArr,
                                      W2, scale2, shift2, (int*)d_out);
}
